// RNNLayer_33406255628925
// MI455X (gfx1250) — compile-verified, matrix operand order checked by reading, not run on hardware
//
#include <hip/hip_runtime.h>
#include <math.h>

// ---------------- types ----------------
typedef __bf16 bf16_t;
typedef __attribute__((ext_vector_type(16))) __bf16 v16bf;
typedef __attribute__((ext_vector_type(8)))  float  v8f;

union FragBF { v16bf v; uint4 q[2]; };
union AccF   { v8f v; float f[8]; };

// ---------------- problem constants ----------------
#define BB   4
#define TT   4096
#define HH   2048
#define RR   2048
#define II   8192
#define MM   (BB * TT)      // 16384 tokens

// ---------------- GEMM tile config ----------------
#define BM 128
#define BN 128
#define BK 32
#define LDSS 40             // LDS row stride in halves (80B, conflict-free, 16B aligned)

// ============================================================
// f32 -> bf16 conversion (weights / staging)
// ============================================================
__global__ void f32_to_bf16_kernel(const float* __restrict__ in,
                                   bf16_t* __restrict__ out, size_t n) {
    size_t i = (size_t)blockIdx.x * blockDim.x + threadIdx.x;
    if (i < n) out[i] = (bf16_t)in[i];
}

// ============================================================
// RMSNorm: one block per token, 256 threads reduce H=2048
// ============================================================
template<bool BF16OUT>
__global__ __launch_bounds__(256) void rmsnorm_kernel(
    const float* __restrict__ x, const float* __restrict__ w,
    float* __restrict__ yf, bf16_t* __restrict__ yb)
{
    __shared__ float red[256];
    size_t token = blockIdx.x;
    const float* xp = x + token * HH;
    float s = 0.f;
    for (int i = threadIdx.x; i < HH; i += 256) { float v = xp[i]; s += v * v; }
    red[threadIdx.x] = s;
    __syncthreads();
    for (int off = 128; off > 0; off >>= 1) {
        if ((int)threadIdx.x < off) red[threadIdx.x] += red[threadIdx.x + off];
        __syncthreads();
    }
    float scale = rsqrtf(red[0] * (1.0f / (float)HH) + 1e-6f);
    for (int i = threadIdx.x; i < HH; i += 256) {
        float v = xp[i] * scale * w[i];
        if (BF16OUT) yb[token * HH + i] = (bf16_t)v;
        else         yf[token * HH + i] = v;
    }
}

// ============================================================
// Causal depthwise conv (K=4) + residual on normed x, bf16 out
// ============================================================
__global__ __launch_bounds__(256) void conv_kernel(
    const float* __restrict__ xn, const float* __restrict__ cw,
    const float* __restrict__ cb, bf16_t* __restrict__ xc)
{
    size_t gid = (size_t)blockIdx.x * blockDim.x + threadIdx.x;
    if (gid >= (size_t)MM * HH) return;
    int d = (int)(gid % HH);
    size_t token = gid / HH;
    int t = (int)(token % TT);
    float acc = xn[gid] + cb[d];
#pragma unroll
    for (int i = 0; i < 4; ++i) {
        int tt = t - 3 + i;
        if (tt >= 0)
            acc += xn[(token - (size_t)(3 - i)) * HH + d] * cw[i * HH + d];
    }
    xc[gid] = (bf16_t)acc;
}

// ============================================================
// Gated linear scan: one thread per (b,r) channel, serial over T
// ============================================================
__global__ __launch_bounds__(256) void scan_kernel(
    const float* __restrict__ zh, bf16_t* __restrict__ h_o)
{
    int idx = blockIdx.x * blockDim.x + threadIdx.x;   // 0 .. B*R-1
    int b = idx / RR, r = idx % RR;
    const float* zp = zh + (size_t)b * TT * (2 * RR) + r;
    float h = 0.f;
#pragma unroll 4
    for (int t = 0; t < TT; ++t) {
        float zt = zp[(size_t)t * (2 * RR)];
        float ht = zp[(size_t)t * (2 * RR) + RR];
        float z  = 1.0f / (1.0f + expf(-zt));
        h = (1.0f - z) * h + z * ht;
        h_o[((size_t)b * TT + t) * RR + r] = (bf16_t)h;
    }
}

// ============================================================
// bf16 WMMA GEMM: C[M,N] = A[M,K] * W[N,K]^T  (f32 accumulate)
// block 128x128, 8 waves, wave tile 32x64 (2x4 of 16x16)
// Double-buffered LDS fed by global_load_async_to_lds_b128 (ASYNCcnt),
// peeled pipeline: steady-state loop is branch-free.
// EPI: 0 -> f32 store; 1 -> f32 + Resid add; 2 -> exact GELU, bf16 store
// ============================================================
#define ASYNC_CP_B128(ldsaddr, gaddr)                                        \
    asm volatile("global_load_async_to_lds_b128 %0, %1, off"                 \
                 :: "v"(ldsaddr), "v"(gaddr) : "memory")

template<int EPI>
__global__ __launch_bounds__(256) void gemm_bf16_wmma_kernel(
    const bf16_t* __restrict__ A, const bf16_t* __restrict__ W,
    float* __restrict__ Cout, bf16_t* __restrict__ CoutBf,
    const float* __restrict__ Resid, int M, int N, int K)
{
    __shared__ __align__(16) bf16_t As[2][BM * LDSS];
    __shared__ __align__(16) bf16_t Ws[2][BN * LDSS];

    int tid  = threadIdx.x;
    int lane = tid & 31;
    int wave = tid >> 5;
    int waveM = wave >> 1;          // 0..3
    int waveN = wave & 1;           // 0..1
    int bm = blockIdx.y * BM;
    int bn = blockIdx.x * BN;

    AccF acc[2][4];
#pragma unroll
    for (int i = 0; i < 2; ++i)
#pragma unroll
        for (int j = 0; j < 4; ++j)
#pragma unroll
            for (int e = 0; e < 8; ++e) acc[i][j].f[e] = 0.f;

    // ---- per-thread staging addresses: 2 chunks of A, 2 of W (16B each) ----
    int rA0 = tid >> 2;             // rows 0..63
    int rA1 = rA0 + 64;             // rows 64..127
    int cA  = (tid & 3) << 3;       // 0,8,16,24 halves within the K-block

    uint64_t gA0 = (uint64_t)(uintptr_t)(A + (size_t)(bm + rA0) * K + cA);
    uint64_t gA1 = (uint64_t)(uintptr_t)(A + (size_t)(bm + rA1) * K + cA);
    uint64_t gW0 = (uint64_t)(uintptr_t)(W + (size_t)(bn + rA0) * K + cA);
    uint64_t gW1 = (uint64_t)(uintptr_t)(W + (size_t)(bn + rA1) * K + cA);

    uint32_t lA0[2], lA1[2], lW0[2], lW1[2];
#pragma unroll
    for (int c = 0; c < 2; ++c) {
        lA0[c] = (uint32_t)(uintptr_t)&As[c][rA0 * LDSS + cA];
        lA1[c] = (uint32_t)(uintptr_t)&As[c][rA1 * LDSS + cA];
        lW0[c] = (uint32_t)(uintptr_t)&Ws[c][rA0 * LDSS + cA];
        lW1[c] = (uint32_t)(uintptr_t)&Ws[c][rA1 * LDSS + cA];
    }

    int hrow = lane & 15;           // tile row (A/W) or N column (C/D)
    int koff = (lane >> 4) << 3;    // lanes 0-15: K 0..7/16..23 ; 16-31: K 8..15/24..31

    // one K-block of WMMAs out of LDS buffer c
    auto compute_tile = [&](int c) {
        const bf16_t* Ac = As[c];
        const bf16_t* Wc = Ws[c];
        FragBF a[2], b[4];
#pragma unroll
        for (int mt = 0; mt < 2; ++mt) {
            const bf16_t* p = Ac + (waveM * 32 + mt * 16 + hrow) * LDSS + koff;
            a[mt].q[0] = *(const uint4*)(p);
            a[mt].q[1] = *(const uint4*)(p + 16);
        }
#pragma unroll
        for (int nt = 0; nt < 4; ++nt) {
            const bf16_t* p = Wc + (waveN * 64 + nt * 16 + hrow) * LDSS + koff;
            b[nt].q[0] = *(const uint4*)(p);
            b[nt].q[1] = *(const uint4*)(p + 16);
        }
#pragma unroll
        for (int mt = 0; mt < 2; ++mt)
#pragma unroll
            for (int nt = 0; nt < 4; ++nt)
                acc[mt][nt].v = __builtin_amdgcn_wmma_f32_16x16x32_bf16(
                    false, a[mt].v, false, b[nt].v,
                    (short)0, acc[mt][nt].v, false, false);
    };

    // ---- prologue: async-fill buffer 0 with K-block 0 ----
    ASYNC_CP_B128(lA0[0], gA0);
    ASYNC_CP_B128(lA1[0], gA1);
    ASYNC_CP_B128(lW0[0], gW0);
    ASYNC_CP_B128(lW1[0], gW1);
    gA0 += BK * 2; gA1 += BK * 2; gW0 += BK * 2; gW1 += BK * 2;

    // ---- steady state: branch-free, always prefetch next into 'nxt' ----
    int cur = 0;
    for (int k0 = 0; k0 < K - BK; k0 += BK) {
        int nxt = cur ^ 1;
        ASYNC_CP_B128(lA0[nxt], gA0);
        ASYNC_CP_B128(lA1[nxt], gA1);
        ASYNC_CP_B128(lW0[nxt], gW0);
        ASYNC_CP_B128(lW1[nxt], gW1);
        gA0 += BK * 2; gA1 += BK * 2; gW0 += BK * 2; gW1 += BK * 2;
        asm volatile("s_wait_asynccnt 0x4" ::: "memory");
        __syncthreads();            // 'cur' resident for all waves
        compute_tile(cur);
        __syncthreads();            // all waves done reading 'cur' before refill
        cur = nxt;
    }

    // ---- tail: last K-block ----
    asm volatile("s_wait_asynccnt 0x0" ::: "memory");
    __syncthreads();
    compute_tile(cur);

    // ---- epilogue: lane l, vgpr e -> (M = e + 8*(l>=16), N = l&15) ----
    int rhi = (lane >> 4) << 3;
#pragma unroll
    for (int mt = 0; mt < 2; ++mt) {
#pragma unroll
        for (int nt = 0; nt < 4; ++nt) {
            int col  = bn + waveN * 64 + nt * 16 + hrow;
            int rowb = bm + waveM * 32 + mt * 16 + rhi;
#pragma unroll
            for (int e = 0; e < 8; ++e) {
                size_t idx = (size_t)(rowb + e) * N + col;
                float v = acc[mt][nt].f[e];
                if (EPI == 0) {
                    Cout[idx] = v;
                } else if (EPI == 1) {
                    Cout[idx] = v + Resid[idx];
                } else {   // EPI == 2: exact gelu -> bf16
                    float g = 0.5f * v * (1.0f + erff(v * 0.70710678118654752f));
                    CoutBf[idx] = (bf16_t)g;
                }
            }
        }
    }
}

// ============================================================
// launch
// ============================================================
extern "C" void kernel_launch(void* const* d_in, const int* in_sizes, int n_in,
                              void* d_out, int out_size, void* d_ws, size_t ws_size,
                              hipStream_t stream) {
    const float* hidden   = (const float*)d_in[0];   // [B,T,H]
    const float* w_zh     = (const float*)d_in[1];   // [2R,H]
    const float* w_resid  = (const float*)d_in[2];   // [H,R]
    const float* conv_w   = (const float*)d_in[3];   // [K,H]
    const float* conv_b   = (const float*)d_in[4];   // [H]
    const float* mlp_up   = (const float*)d_in[5];   // [I,H]
    const float* mlp_down = (const float*)d_in[6];   // [H,I]
    const float* norm1_w  = (const float*)d_in[7];   // [H]
    const float* norm2_w  = (const float*)d_in[8];   // [H]
    float* out = (float*)d_out;                      // [B,T,H]

    // ---- workspace layout (256B aligned) ----
    char* ws = (char*)d_ws;
    size_t off = 0;
    auto alloc = [&](size_t bytes) { char* p = ws + off; off += (bytes + 255) & ~(size_t)255; return p; };

    bf16_t* w_zh_b     = (bf16_t*)alloc((size_t)2 * RR * HH * 2);   //  16.8 MB
    bf16_t* w_resid_b  = (bf16_t*)alloc((size_t)HH * RR * 2);       //   8.4 MB
    bf16_t* mlp_up_b   = (bf16_t*)alloc((size_t)II * HH * 2);       //  33.6 MB
    bf16_t* mlp_down_b = (bf16_t*)alloc((size_t)HH * II * 2);       //  33.6 MB
    float*  xn         = (float*)alloc((size_t)MM * HH * 4);        // 134 MB (reused as resid2)
    bf16_t* xc         = (bf16_t*)alloc((size_t)MM * HH * 2);       //  67 MB (reused as h_o)
    float*  zh         = (float*)alloc((size_t)MM * 2 * RR * 4);    // 268 MB (reused as g)
    bf16_t* xn2        = (bf16_t*)alloc((size_t)MM * HH * 2);       //  67 MB
    float*  resid2     = xn;            // xn dead after conv
    bf16_t* h_o        = xc;            // xc dead after zh GEMM
    bf16_t* g          = (bf16_t*)zh;   // zh dead after scan

    // ---- 1. weights fp32 -> bf16 ----
    {
        auto cvt = [&](const float* src, bf16_t* dst, size_t n) {
            f32_to_bf16_kernel<<<(unsigned)((n + 255) / 256), 256, 0, stream>>>(src, dst, n);
        };
        cvt(w_zh,     w_zh_b,     (size_t)2 * RR * HH);
        cvt(w_resid,  w_resid_b,  (size_t)HH * RR);
        cvt(mlp_up,   mlp_up_b,   (size_t)II * HH);
        cvt(mlp_down, mlp_down_b, (size_t)HH * II);
    }

    // ---- 2. rmsnorm1 -> xn (f32) ----
    rmsnorm_kernel<false><<<MM, 256, 0, stream>>>(hidden, norm1_w, xn, nullptr);

    // ---- 3. conv + residual -> xc (bf16) ----
    conv_kernel<<<(unsigned)(((size_t)MM * HH + 255) / 256), 256, 0, stream>>>(
        xn, conv_w, conv_b, xc);

    // ---- 4. zh = xc @ w_zh^T  [16384 x 4096] ----
    gemm_bf16_wmma_kernel<0><<<dim3(2 * RR / BN, MM / BM), 256, 0, stream>>>(
        xc, w_zh_b, zh, nullptr, nullptr, MM, 2 * RR, HH);

    // ---- 5. gated scan -> h_o (bf16) ----
    scan_kernel<<<(BB * RR) / 256, 256, 0, stream>>>(zh, h_o);

    // ---- 6. resid2 = hidden + h_o @ w_resid^T ----
    gemm_bf16_wmma_kernel<1><<<dim3(HH / BN, MM / BM), 256, 0, stream>>>(
        h_o, w_resid_b, resid2, nullptr, hidden, MM, HH, RR);

    // ---- 7. rmsnorm2 -> xn2 (bf16) ----
    rmsnorm_kernel<true><<<MM, 256, 0, stream>>>(resid2, norm2_w, nullptr, xn2);

    // ---- 8. g = gelu(xn2 @ mlp_up^T) (bf16)  [16384 x 8192] ----
    gemm_bf16_wmma_kernel<2><<<dim3(II / BN, MM / BM), 256, 0, stream>>>(
        xn2, mlp_up_b, nullptr, g, nullptr, MM, II, HH);

    // ---- 9. out = resid2 + g @ mlp_down^T ----
    gemm_bf16_wmma_kernel<1><<<dim3(HH / BN, MM / BM), 256, 0, stream>>>(
        g, mlp_down_b, out, nullptr, resid2, MM, HH, II);
}
